// MambaBlock_25288767439161
// MI455X (gfx1250) — matrix operand order repaired and checked by reading, not compiled, not run on hardware
//
#include <hip/hip_runtime.h>

// ---------------- constants (match reference) ----------------
#define D_MODEL 1024
#define D_INNER 2048
#define D_STATE 16
#define D_CONV  4
#define DT_RANK 64
#define NDBL    (DT_RANK + 2 * D_STATE)   // 96
#define BATCH   2
#define SEQL    2048
#define MTOK    (BATCH * SEQL)            // 4096

// GEMM tiling
#define BM 256            // block M tile (8 waves x 32 rows)
#define BN 64             // block N tile
#define LDSW 40           // shorts per staged column (80B, 16B-aligned, bank spread)

// ---------------- types ----------------
typedef __attribute__((ext_vector_type(16))) __bf16 v16bf;
typedef __attribute__((ext_vector_type(8)))  float  v8f;
typedef int vi4 __attribute__((vector_size(16)));   // matches builtin param pointee

union Frag { v16bf v; uint4 u[2]; };

// ---------------- async global->LDS path detection ----------------
#if __has_builtin(__builtin_amdgcn_global_load_async_to_lds_b128)
#define HAVE_ASYNC_LDS 1
#else
#define HAVE_ASYNC_LDS 0
#endif

#if HAVE_ASYNC_LDS
__device__ __forceinline__ void async_b128(const unsigned short* g, unsigned short* l) {
    __builtin_amdgcn_global_load_async_to_lds_b128(
        (__attribute__((address_space(1))) vi4*)g,
        (__attribute__((address_space(3))) vi4*)l, 0, 0);
}
__device__ __forceinline__ void wait_async_le1() {
#if __has_builtin(__builtin_amdgcn_s_wait_asynccnt)
    __builtin_amdgcn_s_wait_asynccnt(1);
#else
    asm volatile("s_wait_asynccnt 1" ::: "memory");
#endif
}
__device__ __forceinline__ void wait_async_le0() {
#if __has_builtin(__builtin_amdgcn_s_wait_asynccnt)
    __builtin_amdgcn_s_wait_asynccnt(0);
#else
    asm volatile("s_wait_asynccnt 0" ::: "memory");
#endif
}
#endif

// ---------------- helpers ----------------
__device__ __forceinline__ unsigned short f2bf(float f) {
    unsigned int u = __float_as_uint(f);
    u += 0x7FFFu + ((u >> 16) & 1u);            // round-to-nearest-even
    return (unsigned short)(u >> 16);
}
__device__ __forceinline__ float bf2f(unsigned short h) {
    return __uint_as_float(((unsigned int)h) << 16);
}

// ---------------- fp32 -> bf16 convert ----------------
__global__ void cvt_f32_bf16(const float* __restrict__ in,
                             unsigned short* __restrict__ out, int n) {
    int i = blockIdx.x * blockDim.x + threadIdx.x;
    if (i < n) out[i] = f2bf(in[i]);
}

// ---------------- RMSNorm (one token per block, D_MODEL=1024, 256 thr) -----
__global__ void rmsnorm_bf16(const float* __restrict__ x,
                             const float* __restrict__ w,
                             unsigned short* __restrict__ out) {
    int tok = blockIdx.x;
    int tid = threadIdx.x;
    const float* xr = x + (size_t)tok * D_MODEL;
    float v0 = xr[tid], v1 = xr[tid + 256], v2 = xr[tid + 512], v3 = xr[tid + 768];
    float ss = v0 * v0 + v1 * v1 + v2 * v2 + v3 * v3;
    for (int off = 16; off; off >>= 1) ss += __shfl_xor(ss, off, 32);
    __shared__ float red[8];
    if ((tid & 31) == 0) red[tid >> 5] = ss;
    __syncthreads();
    float tot = red[0] + red[1] + red[2] + red[3] + red[4] + red[5] + red[6] + red[7];
    float sc = rsqrtf(tot / (float)D_MODEL + 1e-5f);
    unsigned short* o = out + (size_t)tok * D_MODEL;
    o[tid]       = f2bf(v0 * sc * w[tid]);
    o[tid + 256] = f2bf(v1 * sc * w[tid + 256]);
    o[tid + 512] = f2bf(v2 * sc * w[tid + 512]);
    o[tid + 768] = f2bf(v3 * sc * w[tid + 768]);
}

// ---------------- bf16 WMMA GEMM with LDS-staged B ------------------------
// C[M,N] = act( A[M,K](bf16) * W[N,K]^T(bf16) + bias + resid )
// mode 0: f32 out (+bias +resid)   mode 1: f32 softplus(+bias)
// mode 2: bf16 silu(+bias)
// Block: 8 waves, 256x64 tile. Wave: 32x64 (2 M-subtiles x 4 N-subtiles).
// B tile staged in LDS (async double-buffered when available).
__global__ void gemm_wmma_bf16(const unsigned short* __restrict__ A,
                               const unsigned short* __restrict__ W,
                               const float* __restrict__ bias,
                               const float* __restrict__ resid,
                               void* __restrict__ outp,
                               int M, int N, int K, int mode) {
    __shared__ unsigned short Bsm[2][BN * LDSW];

    const int lane = threadIdx.x;               // 0..31
    const int wave = threadIdx.y;               // 0..7
    const int tid  = wave * 32 + lane;          // 0..255
    const int m0   = blockIdx.x * BM + wave * 32;
    const int n0   = blockIdx.y * BN;

    const int l15 = lane & 15;
    const int hi  = lane >> 4;                  // half-wave select
    const int kbA = hi * 8;                     // A: K 0..7 / 8..15 split
    const int kbB = hi * 8;                     // B: K 0..7 / 8..15 split

    // staging: each thread moves one 16B chunk (8 bf16) of the B tile per K-step
    const int scol   = tid >> 2;                // 0..63 (local column)
    const int schunk = tid & 3;                 // 0..3  (k = schunk*8 ..)
    int gcol = n0 + scol; if (gcol >= N) gcol = N - 1;
    const unsigned short* Wsrc = W + (size_t)gcol * K + schunk * 8;
    unsigned short* Bdst[2] = { &Bsm[0][scol * LDSW + schunk * 8],
                                &Bsm[1][scol * LDSW + schunk * 8] };

    // A row pointers (two 16-row M-subtiles per wave)
    int ar0 = m0 + l15;      if (ar0 >= M) ar0 = M - 1;
    int ar1 = m0 + 16 + l15; if (ar1 >= M) ar1 = M - 1;
    const unsigned short* Arow0 = A + (size_t)ar0 * K;
    const unsigned short* Arow1 = A + (size_t)ar1 * K;

    v8f acc[2][4] = {};
    const int nsteps = K >> 5;                  // K / 32

#if HAVE_ASYNC_LDS
    // prologue: stage step 0 into buffer 0
    async_b128(Wsrc, Bdst[0]);
#endif

    for (int s = 0; s < nsteps; ++s) {
        const int k0  = s << 5;
        const int buf = s & 1;

        // issue A fragment loads early (in flight across the barrier)
        Frag a0, a1;
        a0.u[0] = *(const uint4*)(Arow0 + k0 + kbA);
        a0.u[1] = *(const uint4*)(Arow0 + k0 + 16 + kbA);
        a1.u[0] = *(const uint4*)(Arow1 + k0 + kbA);
        a1.u[1] = *(const uint4*)(Arow1 + k0 + 16 + kbA);

#if HAVE_ASYNC_LDS
        if (s + 1 < nsteps) {
            async_b128(Wsrc + ((s + 1) << 5), Bdst[buf ^ 1]);  // prefetch next
            wait_async_le1();                                  // current staged
        } else {
            wait_async_le0();
        }
        __syncthreads();                                       // tile visible to all waves
#else
        uint4 t = *(const uint4*)(Wsrc + k0);
        __syncthreads();                                       // prior reads done
        *(uint4*)Bdst[0] = t;
        __syncthreads();                                       // tile visible
#endif

#if HAVE_ASYNC_LDS
        const unsigned short* Bb = &Bsm[buf][0];
#else
        const unsigned short* Bb = &Bsm[0][0];
#endif
        #pragma unroll
        for (int j = 0; j < 4; j++) {
            Frag b;
            const unsigned short* bs = Bb + (j * 16 + l15) * LDSW + kbB;
            b.u[0] = *(const uint4*)(bs);
            b.u[1] = *(const uint4*)(bs + 16);
            acc[0][j] = __builtin_amdgcn_wmma_f32_16x16x32_bf16(
                false, a0.v, false, b.v, (short)0, acc[0][j], false, false);
            acc[1][j] = __builtin_amdgcn_wmma_f32_16x16x32_bf16(
                false, a1.v, false, b.v, (short)0, acc[1][j], false, false);
        }
#if HAVE_ASYNC_LDS
        __syncthreads();                                       // reads done before restage
#endif
    }

    // epilogue
    #pragma unroll
    for (int j = 0; j < 4; j++) {
        int col = n0 + j * 16 + l15;
        if (col >= N) continue;
        float bv = bias ? bias[col] : 0.0f;
        #pragma unroll
        for (int mt = 0; mt < 2; mt++) {
            #pragma unroll
            for (int r = 0; r < 8; r++) {
                int row = m0 + mt * 16 + r + hi * 8;
                if (row >= M) continue;
                float v = acc[mt][j][r] + bv;
                size_t oidx = (size_t)row * N + col;
                if (mode == 0) {
                    float o = v;
                    if (resid) o += resid[oidx];
                    ((float*)outp)[oidx] = o;
                } else if (mode == 1) {
                    ((float*)outp)[oidx] = (v > 20.0f) ? v : log1pf(__expf(v));
                } else {                          // mode 2: silu -> bf16
                    float sgl = v * (1.0f / (1.0f + __expf(-v)));
                    ((unsigned short*)outp)[oidx] = f2bf(sgl);
                }
            }
        }
    }
}

// ---------------- depthwise causal conv (win=4) + SiLU -> bf16 -------------
__global__ void conv_silu(const float* __restrict__ xb,
                          const float* __restrict__ cw,
                          const float* __restrict__ cb,
                          unsigned short* __restrict__ out) {
    size_t i = (size_t)blockIdx.x * blockDim.x + threadIdx.x;
    size_t total = (size_t)MTOK * D_INNER;
    if (i >= total) return;
    int d = (int)(i % D_INNER);
    int l = (int)((i / D_INNER) % SEQL);
    int b = (int)(i / ((size_t)D_INNER * SEQL));
    const float* w = cw + d * 4;
    float s = cb[d];
    #pragma unroll
    for (int j = 0; j < 4; j++) {
        int ll = l - 3 + j;
        if (ll >= 0) s += w[j] * xb[((size_t)b * SEQL + ll) * D_INNER + d];
    }
    float y = s * (1.0f / (1.0f + __expf(-s)));
    out[i] = f2bf(y);
}

// ---------------- pack dt_low cols of x_dbl to bf16 -------------------------
__global__ void pack_dtlow(const float* __restrict__ xdbl,
                           unsigned short* __restrict__ dtlow) {
    int i = blockIdx.x * blockDim.x + threadIdx.x;
    if (i >= MTOK * DT_RANK) return;
    int r = i >> 6, k = i & 63;
    dtlow[i] = f2bf(xdbl[(size_t)r * NDBL + k]);
}

// ---------------- selective scan: lane per (b,d,n) --------------------------
__global__ void selective_scan(const float* __restrict__ dt,
                               const unsigned short* __restrict__ xc,
                               const float* __restrict__ xdbl,
                               const float* __restrict__ A_log,
                               const float* __restrict__ Dp,
                               const unsigned short* __restrict__ zs,
                               unsigned short* __restrict__ yz) {
    int t = blockIdx.x * blockDim.x + threadIdx.x;  // 0 .. B*Di*16-1
    if (t >= BATCH * D_INNER * D_STATE) return;
    int n = t & 15;
    int d = (t >> 4) & (D_INNER - 1);
    int b = t >> 15;

    float Adn = -__expf(A_log[d * D_STATE + n]);
    float Dv  = Dp[d];
    float h = 0.0f;
    for (int l = 0; l < SEQL; l++) {
        size_t idx = ((size_t)(b * SEQL + l)) * D_INNER + d;
        size_t ri  = ((size_t)(b * SEQL + l)) * NDBL;
        float dtv = dt[idx];
        float xv  = bf2f(xc[idx]);
        float Bv  = xdbl[ri + DT_RANK + n];
        float Cv  = xdbl[ri + DT_RANK + D_STATE + n];
        float dA  = __expf(dtv * Adn);
        h = dA * h + dtv * Bv * xv;
        float p = h * Cv;
        p += __shfl_xor(p, 1, 32);
        p += __shfl_xor(p, 2, 32);
        p += __shfl_xor(p, 4, 32);
        p += __shfl_xor(p, 8, 32);
        if (n == 0) {
            float y = p + xv * Dv;
            yz[idx] = f2bf(y * bf2f(zs[idx]));
        }
    }
}

// ---------------- host launch ----------------
extern "C" void kernel_launch(void* const* d_in, const int* in_sizes, int n_in,
                              void* d_out, int out_size, void* d_ws, size_t ws_size,
                              hipStream_t stream) {
    const float* x      = (const float*)d_in[0];
    const float* norm_w = (const float*)d_in[1];
    const float* Wx     = (const float*)d_in[2];
    const float* bx     = (const float*)d_in[3];
    const float* Wz     = (const float*)d_in[4];
    const float* bz     = (const float*)d_in[5];
    const float* conv_w = (const float*)d_in[6];
    const float* conv_b = (const float*)d_in[7];
    const float* Wxp    = (const float*)d_in[8];
    const float* Wdt    = (const float*)d_in[9];
    const float* bdt    = (const float*)d_in[10];
    const float* A_log  = (const float*)d_in[11];
    const float* Dp     = (const float*)d_in[12];
    const float* Wout   = (const float*)d_in[13];
    const float* bout   = (const float*)d_in[14];

    // workspace carve-out (256B aligned)
    char* base = (char*)d_ws;
    size_t off = 0;
    auto carve = [&](size_t bytes) -> void* {
        void* p = base + off;
        off = (off + bytes + 255) & ~(size_t)255;
        return p;
    };
    unsigned short* xn_bf    = (unsigned short*)carve((size_t)MTOK * D_MODEL * 2);
    unsigned short* Wx_bf    = (unsigned short*)carve((size_t)D_INNER * D_MODEL * 2);
    unsigned short* Wz_bf    = (unsigned short*)carve((size_t)D_INNER * D_MODEL * 2);
    unsigned short* Wxp_bf   = (unsigned short*)carve((size_t)NDBL * D_INNER * 2);
    unsigned short* Wdt_bf   = (unsigned short*)carve((size_t)D_INNER * DT_RANK * 2);
    unsigned short* Wout_bf  = (unsigned short*)carve((size_t)D_MODEL * D_INNER * 2);
    float*          xb_f32   = (float*)carve((size_t)MTOK * D_INNER * 4);
    unsigned short* zs_bf    = (unsigned short*)carve((size_t)MTOK * D_INNER * 2);
    unsigned short* xc_bf    = (unsigned short*)carve((size_t)MTOK * D_INNER * 2);
    float*          xdbl_f32 = (float*)carve((size_t)MTOK * NDBL * 4);
    unsigned short* dtlow_bf = (unsigned short*)carve((size_t)MTOK * DT_RANK * 2);
    float*          dt_f32   = (float*)carve((size_t)MTOK * D_INNER * 4);
    unsigned short* yz_bf    = (unsigned short*)carve((size_t)MTOK * D_INNER * 2);
    (void)ws_size; (void)n_in; (void)in_sizes; (void)out_size;

    // 1) weight converts
    auto cvt = [&](const float* src, unsigned short* dst, int n) {
        cvt_f32_bf16<<<(n + 255) / 256, 256, 0, stream>>>(src, dst, n);
    };
    cvt(Wx,   Wx_bf,   D_INNER * D_MODEL);
    cvt(Wz,   Wz_bf,   D_INNER * D_MODEL);
    cvt(Wxp,  Wxp_bf,  NDBL * D_INNER);
    cvt(Wdt,  Wdt_bf,  D_INNER * DT_RANK);
    cvt(Wout, Wout_bf, D_MODEL * D_INNER);

    // 2) rmsnorm -> bf16 activations
    rmsnorm_bf16<<<MTOK, 256, 0, stream>>>(x, norm_w, xn_bf);

    auto gemm = [&](const unsigned short* A, const unsigned short* W,
                    const float* bias, const float* resid, void* out,
                    int M, int N, int K, int mode) {
        dim3 blk(32, 8);
        dim3 grd((M + BM - 1) / BM, (N + BN - 1) / BN);
        gemm_wmma_bf16<<<grd, blk, 0, stream>>>(A, W, bias, resid, out, M, N, K, mode);
    };

    // 3) x-branch and gate projections
    gemm(xn_bf, Wx_bf, bx, nullptr, xb_f32, MTOK, D_INNER, D_MODEL, 0);
    gemm(xn_bf, Wz_bf, bz, nullptr, zs_bf,  MTOK, D_INNER, D_MODEL, 2);  // silu fused

    // 4) depthwise causal conv + silu
    {
        size_t total = (size_t)MTOK * D_INNER;
        conv_silu<<<(unsigned)((total + 255) / 256), 256, 0, stream>>>(xb_f32, conv_w, conv_b, xc_bf);
    }

    // 5) x_dbl = xconv @ Wxp^T   (N=96: dt_low | Bt | Ct)
    gemm(xc_bf, Wxp_bf, nullptr, nullptr, xdbl_f32, MTOK, NDBL, D_INNER, 0);
    pack_dtlow<<<(MTOK * DT_RANK + 255) / 256, 256, 0, stream>>>(xdbl_f32, dtlow_bf);

    // 6) dt = softplus(dt_low @ Wdt^T + bdt)
    gemm(dtlow_bf, Wdt_bf, bdt, nullptr, dt_f32, MTOK, D_INNER, DT_RANK, 1);

    // 7) selective scan + output gate
    selective_scan<<<(BATCH * D_INNER * D_STATE) / 256, 256, 0, stream>>>(
        dt_f32, xc_bf, xdbl_f32, A_log, Dp, zs_bf, yz_bf);

    // 8) out = yz @ Wout^T + bout + residual
    gemm(yz_bf, Wout_bf, bout, x, (float*)d_out, MTOK, D_MODEL, D_INNER, 0);
}
